// Attention_790273982568
// MI455X (gfx1250) — matrix operand order repaired and checked by reading, not compiled, not run on hardware
//
#include <hip/hip_runtime.h>
#include <stddef.h>
#include <stdint.h>

// ---------- types ----------
typedef __attribute__((ext_vector_type(16))) __bf16 v16bf;
typedef __attribute__((ext_vector_type(8)))  float  v8f;
typedef __attribute__((ext_vector_type(4)))  unsigned int v4u;
typedef __attribute__((ext_vector_type(4)))  int    v4i;
typedef __attribute__((ext_vector_type(8)))  int    v8i;

union Frag16 { v16bf v; uint4 q[2]; };
union FragAcc { v8f v; float f[8]; };

#define WMMA_BF16(ACC, AF, BF) \
  (ACC).v = __builtin_amdgcn_wmma_f32_16x16x32_bf16( \
      false, (AF).v, false, (BF).v, (short)0, (ACC).v, false, false)

// ---- async global->LDS (CDNA5, ASYNCcnt) ----
#if defined(__has_builtin)
#  if __has_builtin(__builtin_amdgcn_global_load_async_to_lds_b128) && \
      __has_builtin(__builtin_amdgcn_s_wait_asynccnt)
#    define HAVE_ASYNC_LDS 1
#  endif
#endif

#ifdef HAVE_ASYNC_LDS
// param0: v4i in addrspace(1) (printed as "__device__" by clang in HIP mode),
// param1: v4i in addrspace(3), then imm offset, imm cpol.
#define GAS __attribute__((address_space(1)))
#define LAS __attribute__((address_space(3)))
#define ASYNC_CP16(dst_lds, src_glob) \
  __builtin_amdgcn_global_load_async_to_lds_b128( \
      (GAS v4i*)(src_glob), (LAS v4i*)(dst_lds), 0, 0)
#endif

// =====================================================================
// GEMM:  C[M,N] = A[M,K] * B[N,K]^T   (bf16 in, f32 accum, bf16 out)
// 256 thr = 8 waves, tile 128x128, BK=32, wave tile 32x64.
// LDS ping-pong double buffer; next tile staged via async-load-to-LDS
// (no VGPR round trip) while the current tile's 8 WMMAs execute.
// =====================================================================
__global__ __launch_bounds__(256, 1)
void gemm_nt_bf16(const __bf16* __restrict__ A, const __bf16* __restrict__ B,
                  __bf16* __restrict__ C, int M, int N, int K)
{
  __shared__ __align__(16) __bf16 As[2][128][40];  // 32 + 8 pad (80B stride)
  __shared__ __align__(16) __bf16 Bs[2][128][40];

  const int tid  = threadIdx.x;
  const int lane = tid & 31;
  const int wid  = tid >> 5;
  const int idx  = lane & 15;
  const int half = lane >> 4;

  const int bm = blockIdx.y * 128;
  const int bn = blockIdx.x * 128;
  const int wm = (wid & 3) * 32;
  const int wn = (wid >> 2) * 64;

  FragAcc acc[2][4];
  for (int mi = 0; mi < 2; ++mi)
    for (int ni = 0; ni < 4; ++ni)
      for (int r = 0; r < 8; ++r) acc[mi][ni].f[r] = 0.0f;

  // staging slots: 128x32 bf16 = 512 uint4 per matrix -> 2 per thread
  int srow[2], scol[2];
  for (int i2 = 0; i2 < 2; ++i2) {
    const int s = tid * 2 + i2;
    srow[i2] = s >> 2;
    scol[i2] = (s & 3) * 8;
  }

  // ---- stage tile 0 into buffer 0 ----
#ifdef HAVE_ASYNC_LDS
  for (int i2 = 0; i2 < 2; ++i2) {
    ASYNC_CP16(&As[0][srow[i2]][scol[i2]], A + (size_t)(bm + srow[i2]) * K + scol[i2]);
    ASYNC_CP16(&Bs[0][srow[i2]][scol[i2]], B + (size_t)(bn + srow[i2]) * K + scol[i2]);
  }
  __builtin_amdgcn_s_wait_asynccnt(0);
#else
  for (int i2 = 0; i2 < 2; ++i2) {
    *reinterpret_cast<uint4*>(&As[0][srow[i2]][scol[i2]]) =
        *reinterpret_cast<const uint4*>(A + (size_t)(bm + srow[i2]) * K + scol[i2]);
    *reinterpret_cast<uint4*>(&Bs[0][srow[i2]][scol[i2]]) =
        *reinterpret_cast<const uint4*>(B + (size_t)(bn + srow[i2]) * K + scol[i2]);
  }
#endif
  __syncthreads();

  const int nkt = K >> 5;
  for (int kt = 0; kt < nkt; ++kt) {
    const int cur = kt & 1;
    const int nxt = cur ^ 1;
    const bool more = (kt + 1) < nkt;
    const int k1 = (kt + 1) << 5;

#ifdef HAVE_ASYNC_LDS
    if (more) {   // DMA next tile straight into the other LDS buffer
      for (int i2 = 0; i2 < 2; ++i2) {
        ASYNC_CP16(&As[nxt][srow[i2]][scol[i2]], A + (size_t)(bm + srow[i2]) * K + k1 + scol[i2]);
        ASYNC_CP16(&Bs[nxt][srow[i2]][scol[i2]], B + (size_t)(bn + srow[i2]) * K + k1 + scol[i2]);
      }
    }
#else
    uint4 ra[2], rb[2];
    if (more) {   // overlap the global loads with the WMMAs below
      for (int i2 = 0; i2 < 2; ++i2) {
        ra[i2] = *reinterpret_cast<const uint4*>(A + (size_t)(bm + srow[i2]) * K + k1 + scol[i2]);
        rb[i2] = *reinterpret_cast<const uint4*>(B + (size_t)(bn + srow[i2]) * K + k1 + scol[i2]);
      }
    }
#endif

    // compute current tile: 8 WMMAs
    Frag16 af[2], bfr[4];
    for (int mi = 0; mi < 2; ++mi) {
      const __bf16* ap = &As[cur][wm + mi * 16 + idx][0];
      af[mi].q[0] = *reinterpret_cast<const uint4*>(ap + half * 8);
      af[mi].q[1] = *reinterpret_cast<const uint4*>(ap + 16 + half * 8);
    }
    for (int ni = 0; ni < 4; ++ni) {
      const __bf16* bp = &Bs[cur][wn + ni * 16 + idx][0];
      bfr[ni].q[0] = *reinterpret_cast<const uint4*>(bp + half * 8);
      bfr[ni].q[1] = *reinterpret_cast<const uint4*>(bp + 16 + half * 8);
    }
    for (int mi = 0; mi < 2; ++mi)
      for (int ni = 0; ni < 4; ++ni)
        WMMA_BF16(acc[mi][ni], af[mi], bfr[ni]);

#ifdef HAVE_ASYNC_LDS
    if (more) __builtin_amdgcn_s_wait_asynccnt(0);
#else
    if (more) {
      for (int i2 = 0; i2 < 2; ++i2) {
        *reinterpret_cast<uint4*>(&As[nxt][srow[i2]][scol[i2]]) = ra[i2];
        *reinterpret_cast<uint4*>(&Bs[nxt][srow[i2]][scol[i2]]) = rb[i2];
      }
    }
#endif
    __syncthreads();
  }

  for (int mi = 0; mi < 2; ++mi)
    for (int ni = 0; ni < 4; ++ni)
      for (int r = 0; r < 8; ++r) {
        const int m  = bm + wm + mi * 16 + half * 8 + r;
        const int nn = bn + wn + ni * 16 + idx;
        C[(size_t)m * N + nn] = (__bf16)acc[mi][ni].f[r];
      }
}

// =====================================================================
// Per-head LayerNorm + RoPE (+ V copy). grid = (T, 48): 32 q, 8 k, 8 v
// =====================================================================
__global__ __launch_bounds__(128)
void lnrope_kernel(const __bf16* __restrict__ xqkv,
                   const float* __restrict__ qw, const float* __restrict__ qb,
                   const float* __restrict__ kw, const float* __restrict__ kb,
                   __bf16* __restrict__ qr, __bf16* __restrict__ kr,
                   __bf16* __restrict__ vv)
{
  const int t  = blockIdx.x;
  const int hh = blockIdx.y;
  const int d  = threadIdx.x;
  const __bf16* src = xqkv + (size_t)t * 6144 + (size_t)hh * 128;

  if (hh >= 40) {
    vv[((size_t)t * 8 + (hh - 40)) * 128 + d] = src[d];
    return;
  }

  __shared__ float sbuf[128];
  __shared__ float lnv[128];

  const float x = (float)src[d];
  sbuf[d] = x;
  __syncthreads();
  for (int s = 64; s > 0; s >>= 1) { if (d < s) sbuf[d] += sbuf[d + s]; __syncthreads(); }
  const float mean = sbuf[0] * (1.0f / 128.0f);
  __syncthreads();
  const float dv = x - mean;
  sbuf[d] = dv * dv;
  __syncthreads();
  for (int s = 64; s > 0; s >>= 1) { if (d < s) sbuf[d] += sbuf[d + s]; __syncthreads(); }
  const float var = sbuf[0] * (1.0f / 128.0f);

  const bool isq = hh < 32;
  const float* w = isq ? qw : kw;
  const float* b = isq ? qb : kb;
  const float ln = dv * rsqrtf(var + 1e-5f) * w[d] + b[d];
  lnv[d] = ln;
  __syncthreads();

  const int   i   = d >> 1;
  const float inv = __powf(10000.0f, -(float)i * (1.0f / 64.0f));
  const float ang = (float)t * inv;
  float sn, cs;
  __sincosf(ang, &sn, &cs);
  const float xe = lnv[2 * i];
  const float xo = lnv[2 * i + 1];
  const float res = (d & 1) ? (xe * sn + xo * cs) : (xe * cs - xo * sn);

  if (isq) qr[((size_t)t * 32 + hh) * 128 + d] = (__bf16)res;
  else     kr[((size_t)t * 8 + (hh - 32)) * 128 + d] = (__bf16)res;
}

// =====================================================================
// Flash attention (causal, GQA 32q/8kv, HD=128).
// Block = 1 q-head x 128 q rows (8 waves x 16 rows). KV blocks of 32.
// K tile staged by the Tensor Data Mover; V staged+transposed by threads.
// =====================================================================
__global__ __launch_bounds__(256, 1)
void attn_kernel(const __bf16* __restrict__ qr, const __bf16* __restrict__ kr,
                 const __bf16* __restrict__ vv, __bf16* __restrict__ oo)
{
  __shared__ __align__(16) __bf16 Ks[32][136];     // K tile, +8 pad
  __shared__ __align__(16) __bf16 Vt[128][40];     // V^T tile, +8 pad
  __shared__ __align__(16) __bf16 Ps[8][16][40];   // per-wave P relayout

  const int tid  = threadIdx.x;
  const int lane = tid & 31;
  const int wid  = tid >> 5;
  const int idx  = lane & 15;
  const int half = lane >> 4;
  const int h    = blockIdx.y;
  const int n    = h >> 2;
  const int q0   = blockIdx.x * 128;

  Frag16 qf[4];
  {
    const int qrow = q0 + wid * 16 + idx;
    const __bf16* qptr = qr + ((size_t)qrow * 32 + h) * 128;
    for (int kc = 0; kc < 4; ++kc) {
      qf[kc].q[0] = *reinterpret_cast<const uint4*>(qptr + kc * 32 + half * 8);
      qf[kc].q[1] = *reinterpret_cast<const uint4*>(qptr + kc * 32 + 16 + half * 8);
    }
  }

#if __has_builtin(__builtin_amdgcn_tensor_load_to_lds)
  // TDM descriptor: 2D tile 128x32 bf16, row stride 1024 elems, LDS pad
  // 4 dwords after every 64 dwords -> 136-element LDS row stride.
  v4u g0; v8i g1; v4i g2, g3;
  const unsigned ldsK = (unsigned)(size_t)(&Ks[0][0]);
  g0[0] = 1u;
  g0[1] = ldsK;
  g0[2] = 0u;
  g0[3] = 0u;
  g1[0] = (int)((1u << 16) | (1u << 20) | (5u << 22) | (3u << 25));
  g1[1] = (int)(128u << 16);
  g1[2] = (int)(2048u << 16);
  g1[3] = (int)(128u << 16);
  g1[4] = (int)(32u);
  g1[5] = (int)(1024u);
  g1[6] = 0; g1[7] = 0;
  g2[0] = g2[1] = g2[2] = g2[3] = 0;
  g3[0] = g3[1] = g3[2] = g3[3] = 0;
#endif

  float   mrow[8], lrow[8];
  FragAcc of[8];
  for (int r = 0; r < 8; ++r) { mrow[r] = -__builtin_inff(); lrow[r] = 0.0f; }
  for (int dc = 0; dc < 8; ++dc)
    for (int r = 0; r < 8; ++r) of[dc].f[r] = 0.0f;

  const int nblk  = 4 * (blockIdx.x + 1);
  const int qtime = q0 + wid * 16 + half * 8;

  for (int j = 0; j < nblk; ++j) {
    const int kv0 = j * 32;
    __syncthreads();

#if __has_builtin(__builtin_amdgcn_tensor_load_to_lds)
    if (wid == 0) {
      const uint64_t ga = (uint64_t)(size_t)(kr + (((size_t)kv0) * 8 + n) * 128);
      g0[2] = (unsigned)(ga & 0xFFFFFFFFu);
      g0[3] = (unsigned)((ga >> 32) & 0x1FFFFFFu) | (2u << 30);
#if __clang_major__ >= 23
      v8i g4; g4[0]=g4[1]=g4[2]=g4[3]=g4[4]=g4[5]=g4[6]=g4[7]=0;
      __builtin_amdgcn_tensor_load_to_lds(g0, g1, g2, g3, g4, 0);
#else
      __builtin_amdgcn_tensor_load_to_lds(g0, g1, g2, g3, 0);
#endif
    }
#else
    for (int i2 = 0; i2 < 2; ++i2) {
      const int s = tid * 2 + i2;
      const int row = s >> 4;
      const int c = (s & 15) * 8;
      *reinterpret_cast<uint4*>(&Ks[row][c]) =
          *reinterpret_cast<const uint4*>(kr + (((size_t)(kv0 + row)) * 8 + n) * 128 + c);
    }
#endif

    // stage V tile transposed
    for (int i2 = 0; i2 < 2; ++i2) {
      const int s = tid * 2 + i2;
      const int row = s >> 4;
      const int c = (s & 15) * 8;
      union { uint4 u; __bf16 e[8]; } tmp;
      tmp.u = *reinterpret_cast<const uint4*>(vv + (((size_t)(kv0 + row)) * 8 + n) * 128 + c);
      for (int e2 = 0; e2 < 8; ++e2) Vt[c + e2][row] = tmp.e[e2];
    }

#if __has_builtin(__builtin_amdgcn_tensor_load_to_lds)
    if (wid == 0) __builtin_amdgcn_s_wait_tensorcnt(0);
#endif
    __syncthreads();

    FragAcc sa[2];
    for (int t2 = 0; t2 < 2; ++t2) {
      for (int r = 0; r < 8; ++r) sa[t2].f[r] = 0.0f;
      for (int kc = 0; kc < 4; ++kc) {
        Frag16 bf;
        const __bf16* kp = &Ks[t2 * 16 + idx][kc * 32];
        bf.q[0] = *reinterpret_cast<const uint4*>(kp + half * 8);
        bf.q[1] = *reinterpret_cast<const uint4*>(kp + 16 + half * 8);
        WMMA_BF16(sa[t2], qf[kc], bf);
      }
    }

    for (int t2 = 0; t2 < 2; ++t2)
      for (int r = 0; r < 8; ++r) {
        float s = sa[t2].f[r] * 0.08838834764831845f;
        if (kv0 + t2 * 16 + idx > qtime + r) s = -__builtin_inff();
        sa[t2].f[r] = s;
      }

    float fac[8];
    for (int r = 0; r < 8; ++r) {
      float mx = fmaxf(sa[0].f[r], sa[1].f[r]);
      for (int m2 = 8; m2 >= 1; m2 >>= 1) mx = fmaxf(mx, __shfl_xor(mx, m2, 32));
      const float mn = fmaxf(mrow[r], mx);
      const float fc = __expf(mrow[r] - mn);
      const float p0 = __expf(sa[0].f[r] - mn);
      const float p1 = __expf(sa[1].f[r] - mn);
      sa[0].f[r] = p0;
      sa[1].f[r] = p1;
      float ps = p0 + p1;
      for (int m2 = 8; m2 >= 1; m2 >>= 1) ps += __shfl_xor(ps, m2, 32);
      lrow[r] = lrow[r] * fc + ps;
      mrow[r] = mn;
      fac[r]  = fc;
    }

    for (int dc = 0; dc < 8; ++dc)
      for (int r = 0; r < 8; ++r) of[dc].f[r] *= fac[r];

    for (int t2 = 0; t2 < 2; ++t2)
      for (int r = 0; r < 8; ++r)
        Ps[wid][half * 8 + r][t2 * 16 + idx] = (__bf16)sa[t2].f[r];

    Frag16 pf;
    {
      const __bf16* pp = &Ps[wid][idx][0];
      pf.q[0] = *reinterpret_cast<const uint4*>(pp + half * 8);
      pf.q[1] = *reinterpret_cast<const uint4*>(pp + 16 + half * 8);
    }

    for (int dc = 0; dc < 8; ++dc) {
      Frag16 bf;
      const __bf16* vp = &Vt[dc * 16 + idx][0];
      bf.q[0] = *reinterpret_cast<const uint4*>(vp + half * 8);
      bf.q[1] = *reinterpret_cast<const uint4*>(vp + 16 + half * 8);
      WMMA_BF16(of[dc], pf, bf);
    }
  }

  for (int dc = 0; dc < 8; ++dc)
    for (int r = 0; r < 8; ++r) {
      const int tt = qtime + r;
      const int d  = dc * 16 + idx;
      oo[((size_t)tt * 32 + h) * 128 + d] = (__bf16)(of[dc].f[r] / lrow[r]);
    }
}

// =====================================================================
// launch
// =====================================================================
extern "C" void kernel_launch(void* const* d_in, const int* in_sizes, int n_in,
                              void* d_out, int out_size, void* d_ws, size_t ws_size,
                              hipStream_t stream) {
  (void)in_sizes; (void)n_in; (void)out_size; (void)ws_size;

  const __bf16* x    = (const __bf16*)d_in[0];
  const __bf16* wqkv = (const __bf16*)d_in[3];
  const __bf16* wo   = (const __bf16*)d_in[4];
  const float*  qw   = (const float*)d_in[5];
  const float*  qb   = (const float*)d_in[6];
  const float*  kw   = (const float*)d_in[7];
  const float*  kb   = (const float*)d_in[8];
  __bf16* out = (__bf16*)d_out;

  char* ws = (char*)d_ws;
  const size_t T = 2048;
  __bf16* xqkv = (__bf16*)(ws);
  __bf16* qrb  = (__bf16*)(ws + T * 6144 * 2);
  __bf16* krb  = (__bf16*)(ws + T * 6144 * 2 + T * 4096 * 2);
  __bf16* vvb  = (__bf16*)(ws + T * 6144 * 2 + T * 4096 * 2 + T * 1024 * 2);
  __bf16* aob  = (__bf16*)(ws + T * 6144 * 2 + T * 4096 * 2 + T * 1024 * 4);

  gemm_nt_bf16<<<dim3(48, 16), 256, 0, stream>>>(x, wqkv, xqkv, 2048, 6144, 4096);
  lnrope_kernel<<<dim3(2048, 48), 128, 0, stream>>>(xqkv, qw, qb, kw, kb, qrb, krb, vvb);
  attn_kernel<<<dim3(16, 32), 256, 0, stream>>>(qrb, krb, vvb, aob);
  gemm_nt_bf16<<<dim3(32, 16), 256, 0, stream>>>(aob, wo, out, 2048, 4096, 4096);
}
